// Attention_28930899706081
// MI455X (gfx1250) — matrix operand order repaired and checked by reading, not compiled, not run on hardware
//
#include <hip/hip_runtime.h>
#include <hip/hip_bf16.h>
#include <math.h>

typedef __attribute__((ext_vector_type(16))) _Float16 v16h;
typedef __attribute__((ext_vector_type(8)))  float    v8f;
typedef __attribute__((ext_vector_type(4)))  float    v4f;

#define NPIX 4096   // h*w = 64*64
#define NB   2      // batch
#define DCH  128    // projected channels
#define CCH  256    // input channels
#define EPS  1e-12f

// ---------------------------------------------------------------------------
// Kernel 1: 1x1-conv projection + channelwise L2 normalize -> fp16 in ws.
// grid = (8192/8 pixels, 3 maps), block = 256 (8 wave32, one pixel per wave).
// map 0 = q (from dmap), map 1 = k1 (from fmap1), map 2 = k2 (from fmap2).
// proj layout: [map][batch][pix][DCH] fp16.
// ---------------------------------------------------------------------------
__global__ void proj_norm_kernel(const float* __restrict__ fmap1,
                                 const float* __restrict__ fmap2,
                                 const float* __restrict__ dmap,
                                 const float* __restrict__ Wqk,
                                 _Float16* __restrict__ proj) {
    extern __shared__ float Wt[];                  // [CCH][DCH] transposed, 128 KB
    const int lane = threadIdx.x & 31;
    const int wid  = threadIdx.x >> 5;
    const int mapi = blockIdx.y;

    const float* x = (mapi == 0) ? dmap : ((mapi == 1) ? fmap1 : fmap2);

    // Stage W transposed into LDS (coalesced global read, stride-128 LDS write).
    for (int idx = threadIdx.x; idx < DCH * CCH; idx += 256) {
        int dd = idx >> 8;          // /CCH
        int cc = idx & (CCH - 1);
        Wt[cc * DCH + dd] = Wqk[idx];
    }
    __syncthreads();

    const int g     = blockIdx.x * 8 + wid;        // global pixel 0..8191
    const int batch = g >> 12;
    const int p     = g & (NPIX - 1);
    const float* xp = x + (size_t)batch * CCH * NPIX + p;

    // Each lane computes 4 of the 128 output channels: d = lane + 32u.
    float acc[4] = {0.f, 0.f, 0.f, 0.f};
    for (int c = 0; c < CCH; ++c) {
        float xv = xp[(size_t)c * NPIX];           // uniform per wave -> broadcast
        const float* wr = &Wt[c * DCH];
#pragma unroll
        for (int u = 0; u < 4; ++u)
            acc[u] = fmaf(wr[lane + 32 * u], xv, acc[u]);
    }

    // L2 norm over all 128 channels (4 per lane x 32 lanes).
    float ss = acc[0]*acc[0] + acc[1]*acc[1] + acc[2]*acc[2] + acc[3]*acc[3];
#pragma unroll
    for (int m = 16; m >= 1; m >>= 1)
        ss += __shfl_xor(ss, m, 32);
    float sc = 1.0f / fmaxf(sqrtf(ss), EPS);

    _Float16* op = proj + (((size_t)mapi * NB + batch) * NPIX + p) * DCH;
#pragma unroll
    for (int u = 0; u < 4; ++u)
        op[lane + 32 * u] = (_Float16)(acc[u] * sc);
}

// ---------------------------------------------------------------------------
// Kernel 2: fused  sim = q^T K  (WMMA f16)  +  row softmax  -> d_out.
// grid = (256 query tiles, 2 batches, 2 maps {k1,k2}), block = 256 (8 waves).
// Dynamic LDS: 16 rows x 4096 cols fp32 = 256 KB (needs CDNA5's 320 KB WGP LDS).
// ---------------------------------------------------------------------------
__global__ void attn_softmax_kernel(const _Float16* __restrict__ proj,
                                    float* __restrict__ out) {
    extern __shared__ float smem[];                // [16][4096]
    const int lane  = threadIdx.x & 31;
    const int wid   = threadIdx.x >> 5;
    const int lm    = lane & 15;
    const int grp   = lane >> 4;
    const int qtile = blockIdx.x;                  // 0..255
    const int batch = blockIdx.y;                  // 0..1
    const int mapz  = blockIdx.z;                  // 0 -> k1, 1 -> k2

    const _Float16* qb = proj + ((size_t)batch * NPIX + (size_t)qtile * 16) * DCH;
    const _Float16* kb = proj + (((size_t)(1 + mapz) * NB + batch) * NPIX) * DCH;

    // --- Load A fragments (16x32 fp16 each, K = 0..127 in 4 fragments) ---
    // ISA layout: lanes 0-15 hold M=lane, halves hh<8 -> K=hh, hh>=8 -> K=hh+8;
    //             lanes 16-31 shift K by +8.
    v16h afrag[4];
#pragma unroll
    for (int f = 0; f < 4; ++f) {
#pragma unroll
        for (int hh = 0; hh < 16; ++hh) {
            int kk = hh + ((hh < 8) ? 0 : 8) + 8 * grp;
            afrag[f][hh] = qb[(size_t)lm * DCH + f * 32 + kk];
        }
    }

    // --- Loop over key tiles: each wave owns tiles jt = wid, wid+8, ... ---
    for (int jt = wid; jt < NPIX / 16; jt += 8) {
        const _Float16* kt = kb + (size_t)jt * 16 * DCH;
        // B fragment (32x16): lane = column N (lm), halves hh<8 -> K=hh, hh>=8 -> K=hh+8;
        // lanes 16-31 shift K by +8 (same order as A).
        v16h bfrag[4];
#pragma unroll
        for (int f = 0; f < 4; ++f) {
#pragma unroll
            for (int hh = 0; hh < 16; ++hh) {
                int kk = hh + ((hh < 8) ? 0 : 8) + 8 * grp;
                bfrag[f][hh] = kt[(size_t)lm * DCH + f * 32 + kk];
            }
        }

        v8f acc = {};
#pragma unroll
        for (int f = 0; f < 4; ++f) {
            acc = __builtin_amdgcn_wmma_f32_16x16x32_f16(
                /*neg_a=*/false, afrag[f],
                /*neg_b=*/false, bfrag[f],
                /*c_mod=*/(short)0, acc,
                /*reuse_a=*/false, /*reuse_b=*/false);
        }

        // C/D layout: VGPR r, lane -> (M = r + 8*grp, N = lm). Scatter to LDS.
#pragma unroll
        for (int r = 0; r < 8; ++r)
            smem[(size_t)(r + 8 * grp) * NPIX + jt * 16 + lm] = acc[r];
    }
    __syncthreads();

    // --- Softmax over each 4096-wide row; 2 rows per wave ---
#pragma unroll
    for (int rr = 0; rr < 2; ++rr) {
        const int row = wid * 2 + rr;
        float* srow = smem + (size_t)row * NPIX;
        v4f* srow4 = (v4f*)srow;

        // pass 1: max
        float mx = -INFINITY;
        for (int i = lane; i < NPIX / 4; i += 32) {
            v4f v = srow4[i];
            mx = fmaxf(mx, fmaxf(fmaxf(v.x, v.y), fmaxf(v.z, v.w)));
        }
#pragma unroll
        for (int m = 16; m >= 1; m >>= 1)
            mx = fmaxf(mx, __shfl_xor(mx, m, 32));

        // pass 2: exp + sum (write exp back to LDS)
        float s = 0.f;
        for (int i = lane; i < NPIX / 4; i += 32) {
            v4f v = srow4[i];
            v.x = __expf(v.x - mx);
            v.y = __expf(v.y - mx);
            v.z = __expf(v.z - mx);
            v.w = __expf(v.w - mx);
            srow4[i] = v;
            s += v.x + v.y + v.z + v.w;
        }
#pragma unroll
        for (int m = 16; m >= 1; m >>= 1)
            s += __shfl_xor(s, m, 32);
        float inv = 1.0f / s;

        // pass 3: normalize + coalesced b128 stores to HBM
        size_t obase = (size_t)mapz * ((size_t)NB * NPIX * NPIX)
                     + (size_t)batch * ((size_t)NPIX * NPIX)
                     + (size_t)(qtile * 16 + row) * NPIX;
        v4f* orow4 = (v4f*)(out + obase);
        for (int i = lane; i < NPIX / 4; i += 32) {
            v4f v = srow4[i];
            v.x *= inv; v.y *= inv; v.z *= inv; v.w *= inv;
            orow4[i] = v;
        }
    }
}

// ---------------------------------------------------------------------------
extern "C" void kernel_launch(void* const* d_in, const int* in_sizes, int n_in,
                              void* d_out, int out_size, void* d_ws, size_t ws_size,
                              hipStream_t stream) {
    (void)in_sizes; (void)n_in; (void)out_size; (void)ws_size;
    const float* fmap1 = (const float*)d_in[0];
    const float* fmap2 = (const float*)d_in[1];
    const float* dmap  = (const float*)d_in[2];
    const float* Wqk   = (const float*)d_in[3];
    float* out         = (float*)d_out;
    _Float16* proj     = (_Float16*)d_ws;   // 3 maps * 2 * 4096 * 128 fp16 = 6 MB

    // Stage 1: projections + normalize (8 pixels per block, 3 maps)
    proj_norm_kernel<<<dim3(NPIX * NB / 8, 3), 256,
                       DCH * CCH * sizeof(float), stream>>>(
        fmap1, fmap2, dmap, Wqk, proj);

    // Stage 2: fused WMMA QK^T + softmax (256 KB LDS per workgroup)
    attn_softmax_kernel<<<dim3(NPIX / 16, NB, 2), 256,
                          16 * NPIX * sizeof(float), stream>>>(proj, out);
}
